// DGraph_71725953843339
// MI455X (gfx1250) — matrix operand order repaired and checked by reading, not compiled, not run on hardware
//
#include <hip/hip_runtime.h>
#include <hip/hip_bf16.h>

typedef __bf16 bf16_t;
typedef __attribute__((ext_vector_type(16))) __bf16 v16bf;
typedef __attribute__((ext_vector_type(8)))  __bf16 v8bf;
typedef __attribute__((ext_vector_type(8)))  float  v8f;

#define BB   8
#define CC   128
#define NN   2048
#define OUTC 128
#define NEG_SLOPE 0.2f

__device__ __forceinline__ float leakyf(float v)   { return fmaxf(v, NEG_SLOPE * v); }
__device__ __forceinline__ float sigmoidf(float v) { return 1.0f / (1.0f + __expf(-v)); }

// ---- WMMA fragment loaders (bf16, 16x16x32, wave32) ----
// A (16xK tile, row-major, K contiguous): lane l -> row m=l&15.
// Frag elems 0..7  = K[ko .. ko+7],  elems 8..15 = K[16+ko .. 16+ko+7], ko = 8*(l>>4).
__device__ __forceinline__ v16bf load_frag_a(const bf16_t* base, int ld, int lane) {
    const int m  = lane & 15;
    const int ko = (lane >> 4) << 3;
    const bf16_t* p = base + (size_t)m * ld + ko;
    v8bf c0 = *(const v8bf*)(p);
    v8bf c1 = *(const v8bf*)(p + 16);
    return __builtin_shufflevector(c0, c1, 0,1,2,3,4,5,6,7,8,9,10,11,12,13,14,15);
}
// B (Kx16 tile) loaded from B^T row-major (row n, K contiguous): lane l -> col n=l&15.
// Frag elems 0..7  = K[ko .. ko+7],  elems 8..15 = K[16+ko .. 16+ko+7], ko = 8*(l>>4).
__device__ __forceinline__ v16bf load_frag_b(const bf16_t* baseT, int ld, int lane) {
    const int n  = lane & 15;
    const int ko = (lane >> 4) << 3;
    const bf16_t* p = baseT + (size_t)n * ld + ko;
    v8bf c0 = *(const v8bf*)(p);
    v8bf c1 = *(const v8bf*)(p + 16);
    return __builtin_shufflevector(c0, c1, 0,1,2,3,4,5,6,7,8,9,10,11,12,13,14,15);
}

__device__ __forceinline__ v8f wmma_bf16(v16bf a, v16bf b, v8f c) {
    return __builtin_amdgcn_wmma_f32_16x16x32_bf16(false, a, false, b, (short)0, c, false, false);
}

// One wave computes a (16*MT) x (16*NT) f32 tile.
template <int MT, int NT>
__device__ __forceinline__ void wave_gemm(
    const bf16_t* A, int lda, const bf16_t* BT, int ldb, int K, int lane,
    v8f acc[MT][NT])
{
    for (int k = 0; k < K; k += 32) {
        v16bf fa[MT], fb[NT];
#pragma unroll
        for (int i = 0; i < MT; ++i)
            fa[i] = load_frag_a(A + (size_t)(16 * i) * lda + k, lda, lane);
#pragma unroll
        for (int j = 0; j < NT; ++j)
            fb[j] = load_frag_b(BT + (size_t)(16 * j) * ldb + k, ldb, lane);
#pragma unroll
        for (int i = 0; i < MT; ++i)
#pragma unroll
            for (int j = 0; j < NT; ++j)
                acc[i][j] = wmma_bf16(fa[i], fb[j], acc[i][j]);
    }
}

// C/D tile mapping: vgpr r, lane l -> m = r + 8*(l>>4), n = l&15.
#define EPILOGUE(MT, NT, STMT)                                                   \
    _Pragma("unroll") for (int i = 0; i < (MT); ++i)                             \
    _Pragma("unroll") for (int j = 0; j < (NT); ++j)                             \
    _Pragma("unroll") for (int r = 0; r < 8; ++r) {                              \
        const int mm = m0 + 16 * i + r + ((lane >> 4) << 3);                     \
        const int nn = n0 + 16 * j + (lane & 15);                                \
        const float vv = acc[i][j][r];                                           \
        STMT;                                                                    \
    }

#define WAVE_COORDS(MT, NT)                                                      \
    const int lane = threadIdx.x & 31, wave = threadIdx.x >> 5;                  \
    const int m0 = (blockIdx.x * 2 + (wave >> 1)) * (16 * (MT));                 \
    const int n0 = (blockIdx.y * 2 + (wave & 1)) * (16 * (NT));                  \
    const int b  = blockIdx.z;

// ---- prep: x -> bf16 row-major xb[B][C][N] (coalesced) ----
__global__ __launch_bounds__(256) void k_prep_x(const float* __restrict__ x,
                                                bf16_t* __restrict__ xb) {
    int idx = blockIdx.x * 256 + threadIdx.x;   // B*C*N threads exactly
    xb[idx] = (bf16_t)x[idx];
}

// ---- prep: x^T -> xcT[:, :, 128:256] via 32x32 LDS-staged transpose ----
__global__ __launch_bounds__(256) void k_xpose(const float* __restrict__ x,
                                               bf16_t* __restrict__ xcT) {
    __shared__ bf16_t tile[32][33];
    const int b  = blockIdx.z;
    const int c0 = blockIdx.y * 32;
    const int n0 = blockIdx.x * 32;
    const int t0 = threadIdx.x;   // 0..31
    const int t1 = threadIdx.y;   // 0..7
#pragma unroll
    for (int cc = t1; cc < 32; cc += 8)   // read: coalesced in n
        tile[cc][t0] = (bf16_t)x[((size_t)b * CC + (c0 + cc)) * NN + (n0 + t0)];
    __syncthreads();
#pragma unroll
    for (int nn = t1; nn < 32; nn += 8)   // write: coalesced in c
        xcT[((size_t)b * NN + (n0 + nn)) * (2 * CC) + CC + (c0 + t0)] = tile[t0][nn];
}

__global__ __launch_bounds__(256) void k_cvt(const float* __restrict__ s,
                                             bf16_t* __restrict__ d, int n) {
    int i = blockIdx.x * 256 + threadIdx.x;
    if (i < n) d[i] = (bf16_t)s[i];
}

// ---- GEMM1: energy[b] = xb[b] (CxN) * xb[b]^T -> f32 [C][C] ----
__global__ __launch_bounds__(128) void k_gemm_energy(const bf16_t* __restrict__ xb,
                                                     float* __restrict__ energy) {
    WAVE_COORDS(2, 2);
    const bf16_t* base = xb + (size_t)b * CC * NN;
    v8f acc[2][2] = {};
    wave_gemm<2, 2>(base + (size_t)m0 * NN, NN, base + (size_t)n0 * NN, NN, NN, lane, acc);
    float* E = energy + (size_t)b * CC * CC;
    EPILOGUE(2, 2, E[(size_t)mm * CC + nn] = vv);
}

// ---- softmax(max - energy) == softmax(-energy), per row of 128 -> bf16 att ----
__global__ __launch_bounds__(128) void k_softmax(const float* __restrict__ energy,
                                                 bf16_t* __restrict__ att) {
    const int row = blockIdx.x;     // B*C rows
    const int t   = threadIdx.x;    // 128
    __shared__ float red[128];
    float v = -energy[(size_t)row * CC + t];
    red[t] = v; __syncthreads();
    for (int s = 64; s > 0; s >>= 1) { if (t < s) red[t] = fmaxf(red[t], red[t + s]); __syncthreads(); }
    float m = red[0]; __syncthreads();
    float e = __expf(v - m);
    red[t] = e; __syncthreads();
    for (int s = 64; s > 0; s >>= 1) { if (t < s) red[t] += red[t + s]; __syncthreads(); }
    att[(size_t)row * CC + t] = (bf16_t)(e / red[0]);
}

// ---- GEMM2: out = att (CxC) * x (CxN); epilogue gamma*out + x, transposed store
//      into xcT[:, :, 0:128]. B^T source = xT stored at xcT[:, :, 128:256]. ----
__global__ __launch_bounds__(128) void k_gemm_xglb(const bf16_t* __restrict__ att,
                                                   bf16_t* __restrict__ xcT,
                                                   const float* __restrict__ x,
                                                   const float* __restrict__ gamma) {
    WAVE_COORDS(2, 2);                                   // m0: c, n0: n
    const bf16_t* A  = att + (size_t)b * CC * CC + (size_t)m0 * CC;
    const bf16_t* BT = xcT + (size_t)b * NN * (2 * CC) + (size_t)n0 * (2 * CC) + CC;
    v8f acc[2][2] = {};
    wave_gemm<2, 2>(A, CC, BT, 2 * CC, CC, lane, acc);
    const float g = gamma[0];
    const float*  xB  = x   + (size_t)b * CC * NN;
    bf16_t*       dst = xcT + (size_t)b * NN * (2 * CC);
    EPILOGUE(2, 2,
        dst[(size_t)nn * (2 * CC) + mm] = (bf16_t)(g * vv + xB[(size_t)mm * NN + nn]));
}

// ---- GEMM3 (dominant): adjT[b][n][k] = sigmoid( xcT[b] (Nx2C) * w_adj^T + b_adj[k] )
//      64x64 per wave (16 WMMA per 8 fragment loads). ----
__global__ __launch_bounds__(128) void k_gemm_adj(const bf16_t* __restrict__ xcT,
                                                  const bf16_t* __restrict__ wadj,
                                                  const float* __restrict__ badj,
                                                  bf16_t* __restrict__ adjT) {
    WAVE_COORDS(4, 4);                                   // m0: n (2048), n0: k (2048)
    const bf16_t* A  = xcT + (size_t)b * NN * (2 * CC) + (size_t)m0 * (2 * CC);
    const bf16_t* BT = wadj + (size_t)n0 * (2 * CC);     // w_adj rows are B^T rows
    v8f acc[4][4] = {};
    wave_gemm<4, 4>(A, 2 * CC, BT, 2 * CC, 2 * CC, lane, acc);
    bf16_t* D = adjT + (size_t)b * NN * NN;
    EPILOGUE(4, 4, D[(size_t)mm * NN + nn] = (bf16_t)sigmoidf(vv + badj[nn]));
}

// ---- GEMM4: y1[c][n] = leaky( xb[b] (CxN) * adj ); store transposed y1T[n][c].
//      32x64 per wave, K = 2048 streaming adjT out of L2. ----
__global__ __launch_bounds__(128) void k_gemm_y1(const bf16_t* __restrict__ xb,
                                                 const bf16_t* __restrict__ adjT,
                                                 bf16_t* __restrict__ y1T) {
    WAVE_COORDS(2, 4);                                   // m0: c, n0: n
    const bf16_t* A  = xb   + (size_t)b * CC * NN + (size_t)m0 * NN;
    const bf16_t* BT = adjT + (size_t)b * NN * NN + (size_t)n0 * NN;
    v8f acc[2][4] = {};
    wave_gemm<2, 4>(A, NN, BT, NN, NN, lane, acc);
    bf16_t* D = y1T + (size_t)b * NN * CC;
    EPILOGUE(2, 4, D[(size_t)nn * CC + mm] = (bf16_t)leakyf(vv));
}

// ---- GEMM5: out[b][o][n] = leaky( w_dyn (OUTxC) * y1 + b_dyn[o] ), f32 output ----
__global__ __launch_bounds__(128) void k_gemm_out(const bf16_t* __restrict__ wdyn,
                                                  const bf16_t* __restrict__ y1T,
                                                  const float* __restrict__ bdyn,
                                                  float* __restrict__ out) {
    WAVE_COORDS(2, 2);                                   // m0: o, n0: n
    const bf16_t* A  = wdyn + (size_t)m0 * CC;
    const bf16_t* BT = y1T  + (size_t)b * NN * CC + (size_t)n0 * CC;
    v8f acc[2][2] = {};
    wave_gemm<2, 2>(A, CC, BT, CC, CC, lane, acc);
    float* D = out + (size_t)b * OUTC * NN;
    EPILOGUE(2, 2, D[(size_t)mm * NN + nn] = leakyf(vv + bdyn[mm]));
}

extern "C" void kernel_launch(void* const* d_in, const int* in_sizes, int n_in,
                              void* d_out, int out_size, void* d_ws, size_t ws_size,
                              hipStream_t stream) {
    const float* x     = (const float*)d_in[0];
    const float* w_adj = (const float*)d_in[1];
    const float* b_adj = (const float*)d_in[2];
    const float* w_dyn = (const float*)d_in[3];
    const float* b_dyn = (const float*)d_in[4];
    const float* gamma = (const float*)d_in[5];
    float* out = (float*)d_out;

    char* ws = (char*)d_ws;
    size_t off = 0;
    auto alloc = [&](size_t bytes) { size_t o = off; off += (bytes + 255) & ~(size_t)255; return o; };
    bf16_t* xb     = (bf16_t*)(ws + alloc((size_t)BB * CC * NN * 2));        //  4 MB
    bf16_t* xcT    = (bf16_t*)(ws + alloc((size_t)BB * NN * 2 * CC * 2));    //  8 MB
    bf16_t* wadj   = (bf16_t*)(ws + alloc((size_t)NN * 2 * CC * 2));         //  1 MB
    bf16_t* wdyn   = (bf16_t*)(ws + alloc((size_t)OUTC * CC * 2));           // 32 KB
    float*  energy = (float*) (ws + alloc((size_t)BB * CC * CC * 4));        // 512 KB
    bf16_t* att    = (bf16_t*)(ws + alloc((size_t)BB * CC * CC * 2));        // 256 KB
    bf16_t* adjT   = (bf16_t*)(ws + alloc((size_t)BB * NN * NN * 2));        // 64 MB
    bf16_t* y1T    = (bf16_t*)(ws + alloc((size_t)BB * NN * CC * 2));        //  4 MB

    k_prep_x<<<(BB * CC * NN) / 256, 256, 0, stream>>>(x, xb);
    k_xpose<<<dim3(NN / 32, CC / 32, BB), dim3(32, 8), 0, stream>>>(x, xcT);
    k_cvt<<<(NN * 2 * CC) / 256, 256, 0, stream>>>(w_adj, wadj, NN * 2 * CC);
    k_cvt<<<(OUTC * CC) / 256, 256, 0, stream>>>(w_dyn, wdyn, OUTC * CC);

    k_gemm_energy<<<dim3(CC / 64, CC / 64, BB), 128, 0, stream>>>(xb, energy);
    k_softmax<<<BB * CC, 128, 0, stream>>>(energy, att);
    k_gemm_xglb<<<dim3(CC / 64, NN / 64, BB), 128, 0, stream>>>(att, xcT, x, gamma);
    k_gemm_adj<<<dim3(NN / 128, NN / 128, BB), 128, 0, stream>>>(xcT, wadj, b_adj, adjT);
    k_gemm_y1<<<dim3(CC / 64, NN / 128, BB), 128, 0, stream>>>(xb, adjT, y1T);
    k_gemm_out<<<dim3(OUTC / 64, NN / 64, BB), 128, 0, stream>>>(wdyn, y1T, b_dyn, out);
}
